// SdfDecoder_MultiMLP_82076825026818
// MI455X (gfx1250) — matrix operand order repaired and checked by reading, not compiled, not run on hardware
//
#include <hip/hip_runtime.h>
#include <hip/hip_bf16.h>

typedef __attribute__((ext_vector_type(16))) _Float16 v16h;
typedef __attribute__((ext_vector_type(8)))  _Float16 v8h;
typedef __attribute__((ext_vector_type(2)))  _Float16 v2h;
typedef __attribute__((ext_vector_type(8)))  float    v8f;

#define HID 128
#define SPTS 32768          // S
#define NPTS (16 * 32768)   // B*S = 524288

// ---------------- workspace layout (bytes) ----------------
#define WS_TENC 0
#define WS_B0   1024
#define WS_B1   (1024 + 16384)
#define WS_B2   (1024 + 16384 + 32768)
#define WS_BOUT (1024 + 16384 + 65536)

// ---------------------------------------------------------------------------
// Prep: time-encoding table + weight repack fp32 -> f16 WMMA B-fragments.
// B-fragment layout per 32x16 (KxN) tile: lane l holds column N=(l&15),
// halfword h (0..15) holds K = kt*32 + (l>>4)*16 + h.
// ---------------------------------------------------------------------------
__global__ void sdf_prep(const float* __restrict__ time,
                         const float* __restrict__ te0,
                         const float* __restrict__ te1,
                         const float* __restrict__ W0,
                         const float* __restrict__ W1,
                         const float* __restrict__ W2,
                         const float* __restrict__ Wout,
                         float* __restrict__ tenc,
                         _Float16* __restrict__ B0,
                         _Float16* __restrict__ B1,
                         _Float16* __restrict__ B2,
                         _Float16* __restrict__ Bo)
{
    const int tid = threadIdx.x;

    if (tid < 16) {
        float t = time[tid];
        {   // level 5 encoder -> dims 0..7
            const int L = 5;
            float tL = t * (float)L;
            int idx = (int)floorf(tL);
            idx = idx < 0 ? 0 : (idx > L - 1 ? L - 1 : idx);
            float w = (float)(idx + 1) - tL;
            #pragma unroll
            for (int d = 0; d < 8; ++d) {
                float v = w * te0[idx * 8 + d] + (1.0f - w) * te0[(idx + 1) * 8 + d];
                if (t >= 1.0f) v = te0[L * 8 + d];
                tenc[tid * 16 + d] = v;
            }
        }
        {   // level 20 encoder -> dims 8..15
            const int L = 20;
            float tL = t * (float)L;
            int idx = (int)floorf(tL);
            idx = idx < 0 ? 0 : (idx > L - 1 ? L - 1 : idx);
            float w = (float)(idx + 1) - tL;
            #pragma unroll
            for (int d = 0; d < 8; ++d) {
                float v = w * te1[idx * 8 + d] + (1.0f - w) * te1[(idx + 1) * 8 + d];
                if (t >= 1.0f) v = te1[L * 8 + d];
                tenc[tid * 16 + 8 + d] = v;
            }
        }
    }

    // W0: (55,128) row-major, pad K to 64.  2 kt * 8 nt * 32 * 16 = 8192
    for (int i = tid; i < 2 * 8 * 32 * 16; i += blockDim.x) {
        int h  = i & 15;
        int l  = (i >> 4) & 31;
        int nt = (i >> 9) & 7;
        int kt = i >> 12;
        int K  = kt * 32 + (l >> 4) * 16 + h;
        int n  = nt * 16 + (l & 15);
        float v = (K < 55) ? W0[K * HID + n] : 0.0f;
        B0[i] = (_Float16)v;
    }
    // W1 / W2: (128,128). 4 kt * 8 nt * 32 * 16 = 16384 each
    for (int i = tid; i < 4 * 8 * 32 * 16; i += blockDim.x) {
        int h  = i & 15;
        int l  = (i >> 4) & 31;
        int nt = (i >> 9) & 7;
        int kt = i >> 12;
        int K  = kt * 32 + (l >> 4) * 16 + h;
        int n  = nt * 16 + (l & 15);
        B1[i] = (_Float16)W1[K * HID + n];
        B2[i] = (_Float16)W2[K * HID + n];
    }
    // Wout: (128,1) -> N padded to 16, only column 0 nonzero. 4*32*16 = 2048
    for (int i = tid; i < 4 * 32 * 16; i += blockDim.x) {
        int h  = i & 15;
        int l  = (i >> 4) & 31;
        int kt = i >> 9;
        int K  = kt * 32 + (l >> 4) * 16 + h;
        float v = ((l & 15) == 0) ? Wout[K] : 0.0f;
        Bo[i] = (_Float16)v;
    }
}

// ---------------------------------------------------------------------------
// One 128-wide hidden layer for a 32-point wave slab.
// slab: wave-private LDS, 32 rows x 128 f16 cols.  KT = K/32 tiles.
// A-fragment (16-bit 16x32): lane group g reads K offsets g*16 .. g*16+15.
// nt loop fully unrolled so the scheduler can pipeline the B-fragment
// global loads (L0/L2-resident) behind WMMA issue.
// ---------------------------------------------------------------------------
template<int KT>
__device__ __forceinline__ void mlp_layer(_Float16* slab, int lane,
                                          const _Float16* __restrict__ Bw,
                                          const float* __restrict__ bias)
{
    const int m16 = lane & 15;
    const int g   = lane >> 4;

    // hoist per-lane bias values for all 8 column tiles
    float bs[8];
    #pragma unroll
    for (int nt = 0; nt < 8; ++nt) bs[nt] = bias[nt * 16 + m16];

    v16h A[2][KT];
    #pragma unroll
    for (int mt = 0; mt < 2; ++mt) {
        #pragma unroll
        for (int kt = 0; kt < KT; ++kt) {
            const _Float16* rp = slab + (mt * 16 + m16) * HID + kt * 32 + g * 16;
            v8h lo = *(const v8h*)rp;
            v8h hi = *(const v8h*)(rp + 8);
            A[mt][kt] = __builtin_shufflevector(lo, hi,
                0, 1, 2, 3, 4, 5, 6, 7, 8, 9, 10, 11, 12, 13, 14, 15);
        }
    }

    #pragma unroll
    for (int nt = 0; nt < 8; ++nt) {
        v8f c0 = {};
        v8f c1 = {};
        #pragma unroll
        for (int kt = 0; kt < KT; ++kt) {
            v16h b = *(const v16h*)(Bw + (((size_t)kt * 8 + nt) * 32 + lane) * 16);
            c0 = __builtin_amdgcn_wmma_f32_16x16x32_f16(false, A[0][kt], false, b,
                                                        (short)0, c0, false, false);
            c1 = __builtin_amdgcn_wmma_f32_16x16x32_f16(false, A[1][kt], false, b,
                                                        (short)0, c1, false, false);
        }
        // fused bias + ReLU + f32->f16, back into the slab (D row M = r + 8*g)
        const int col = nt * 16 + m16;
        #pragma unroll
        for (int r = 0; r < 8; ++r) {
            float v0 = c0[r] + bs[nt]; v0 = v0 > 0.0f ? v0 : 0.0f;
            float v1 = c1[r] + bs[nt]; v1 = v1 > 0.0f ? v1 : 0.0f;
            slab[(g * 8 + r) * HID + col]      = (_Float16)v0;
            slab[(16 + g * 8 + r) * HID + col] = (_Float16)v1;
        }
    }
}

// ---------------------------------------------------------------------------
// Main: 256 threads = 8 waves, each wave owns 32 consecutive points.
// ---------------------------------------------------------------------------
struct F3 { float x, y, z; };

__global__ __launch_bounds__(256)
void sdf_main(const float* __restrict__ xyz,
              const float* __restrict__ tenc,
              const _Float16* __restrict__ B0, const float* __restrict__ b0,
              const _Float16* __restrict__ B1, const float* __restrict__ b1,
              const _Float16* __restrict__ B2, const float* __restrict__ b2,
              const _Float16* __restrict__ Bo, const float* __restrict__ bout,
              float* __restrict__ out)
{
    __shared__ _Float16 act[8][32][HID];   // 64 KB: 8 KB wave-private slabs

    const int wave = threadIdx.x >> 5;
    const int lane = threadIdx.x & 31;
    const int base = blockIdx.x * 256 + wave * 32;
    _Float16* slab = &act[wave][0][0];

    // ---- feature stage: each lane builds one point's 64-wide (padded) row ----
    {
        const int p    = base + lane;
        const int bidx = p >> 15;                 // p / S, S = 32768
        _Float16* row  = slab + lane * HID;
        const float* te = tenc + bidx * 16;

        // dims 0..15: time encoding, packed pairs -> ds_store_b32
        #pragma unroll
        for (int d = 0; d < 8; ++d) {
            v2h pr = { (_Float16)te[2 * d], (_Float16)te[2 * d + 1] };
            *(v2h*)(row + 2 * d) = pr;
        }

        F3 q = *(const F3*)(xyz + p * 3);
        float px = (q.x + 1.0f) * 0.5f;
        float py = (q.y + 1.0f) * 0.5f;
        float pz = (q.z + 1.0f) * 0.5f;
        row[16] = (_Float16)px;
        row[17] = (_Float16)py;
        row[18] = (_Float16)pz;

        float f = 1.0f;
        #pragma unroll
        for (int fi = 0; fi < 6; ++fi) {
            row[19 + fi * 3 + 0] = (_Float16)__sinf(px * f);
            row[19 + fi * 3 + 1] = (_Float16)__sinf(py * f);
            row[19 + fi * 3 + 2] = (_Float16)__sinf(pz * f);
            row[37 + fi * 3 + 0] = (_Float16)__cosf(px * f);
            row[37 + fi * 3 + 1] = (_Float16)__cosf(py * f);
            row[37 + fi * 3 + 2] = (_Float16)__cosf(pz * f);
            f *= 2.0f;
        }
        #pragma unroll
        for (int d = 55; d < 64; ++d) row[d] = (_Float16)0.0f;
    }
    // No barrier needed: each wave only touches its own slab; LDS ops from one
    // wave complete in order (DScnt), compiler inserts s_wait_dscnt.

    mlp_layer<2>(slab, lane, B0, b0);   // 55(->64) -> 128
    mlp_layer<4>(slab, lane, B1, b1);   // 128 -> 128
    mlp_layer<4>(slab, lane, B2, b2);   // 128 -> 128

    // ---- output layer: 128 -> 1 via WMMA with N padded to 16 ----
    {
        const int m16 = lane & 15;
        const int g   = lane >> 4;
        v16h A[2][4];
        #pragma unroll
        for (int mt = 0; mt < 2; ++mt) {
            #pragma unroll
            for (int kt = 0; kt < 4; ++kt) {
                const _Float16* rp = slab + (mt * 16 + m16) * HID + kt * 32 + g * 16;
                v8h lo = *(const v8h*)rp;
                v8h hi = *(const v8h*)(rp + 8);
                A[mt][kt] = __builtin_shufflevector(lo, hi,
                    0, 1, 2, 3, 4, 5, 6, 7, 8, 9, 10, 11, 12, 13, 14, 15);
            }
        }
        v8f c0 = {};
        v8f c1 = {};
        #pragma unroll
        for (int kt = 0; kt < 4; ++kt) {
            v16h b = *(const v16h*)(Bo + ((size_t)kt * 32 + lane) * 16);
            c0 = __builtin_amdgcn_wmma_f32_16x16x32_f16(false, A[0][kt], false, b,
                                                        (short)0, c0, false, false);
            c1 = __builtin_amdgcn_wmma_f32_16x16x32_f16(false, A[1][kt], false, b,
                                                        (short)0, c1, false, false);
        }
        if (m16 == 0) {   // only column N==0 carries the result
            const float bo = bout[0];
            #pragma unroll
            for (int r = 0; r < 8; ++r) {
                out[base + g * 8 + r]      = c0[r] + bo;
                out[base + 16 + g * 8 + r] = c1[r] + bo;
            }
        }
    }
}

// ---------------------------------------------------------------------------
extern "C" void kernel_launch(void* const* d_in, const int* in_sizes, int n_in,
                              void* d_out, int out_size, void* d_ws, size_t ws_size,
                              hipStream_t stream) {
    const float* time = (const float*)d_in[0];
    const float* xyz  = (const float*)d_in[1];
    const float* te0  = (const float*)d_in[2];
    const float* te1  = (const float*)d_in[3];
    const float* W0   = (const float*)d_in[4];
    const float* b0   = (const float*)d_in[5];
    const float* W1   = (const float*)d_in[6];
    const float* b1   = (const float*)d_in[7];
    const float* W2   = (const float*)d_in[8];
    const float* b2   = (const float*)d_in[9];
    const float* Wout = (const float*)d_in[10];
    const float* bout = (const float*)d_in[11];

    char* ws = (char*)d_ws;
    float*    tenc = (float*)(ws + WS_TENC);
    _Float16* B0f  = (_Float16*)(ws + WS_B0);
    _Float16* B1f  = (_Float16*)(ws + WS_B1);
    _Float16* B2f  = (_Float16*)(ws + WS_B2);
    _Float16* Bof  = (_Float16*)(ws + WS_BOUT);

    sdf_prep<<<1, 256, 0, stream>>>(time, te0, te1, W0, W1, W2, Wout,
                                    tenc, B0f, B1f, B2f, Bof);

    const int nblocks = NPTS / 256;   // 2048 blocks * 256 points
    sdf_main<<<nblocks, 256, 0, stream>>>(xyz, tenc,
                                          B0f, b0, B1f, b1, B2f, b2, Bof, bout,
                                          (float*)d_out);
}
